// CustonDATransformerLayer_59356448031003
// MI455X (gfx1250) — matrix operand order repaired and checked by reading, not compiled, not run on hardware
//
#include <hip/hip_runtime.h>
#include <hip/hip_bf16.h>

typedef __attribute__((ext_vector_type(16))) _Float16 v16h;
typedef __attribute__((ext_vector_type(8)))  float    v8f;

namespace {
constexpr int EMBED  = 256;
constexpr int HEADS  = 8;
constexpr int LEVELS = 4;
constexpr int POINTS = 4;
constexpr int HD     = 32;
constexpr int BS     = 2;
constexpr int NQ     = 16384;
constexpr int NV     = 128*128 + 64*64 + 32*32 + 16*16;   // 21760
constexpr int MQ     = BS * NQ;                            // 32768
constexpr int MV     = BS * NV;                            // 43520
}

union AFrag { v16h v; unsigned u[8]; };

// ---------------------------------------------------------------------------
// WMMA GEMM: D[M,N] = A[M,K](f16,row-major) * B(f16, fragment-packed) + bias
// Block = 256 threads = 8 waves; wave tile 16(M) x 32(N) (2 accumulators share
// one A fragment); block tile 64(M) x 64(N). K fully unrolled (K/32 WMMAs per
// accumulator). All operands stream straight from global: the entire layer
// (~160MB) is resident in MI455X's 192MB L2, so LDS staging buys nothing here.
//
// Packed-B layout: Bp[((kt*(N/16) + nt)*32 + lane)*16 + j] holds, in ISA
// register order, the 16 halves lane `lane` needs for k-tile kt / n-tile nt:
//   n = nt*16 + (lane&15),  k = kt*32 + (lane<16 ? 0 : 8) + (j<8 ? j : j+8)
// so a B fragment is a single contiguous 32B load per lane.
// ---------------------------------------------------------------------------
template<int HAS_BIAS, int RELU, int OUTF16, int N, int K>
__global__ __launch_bounds__(256)
void gemm_f16_wmma(const _Float16* __restrict__ A, const _Float16* __restrict__ Bp,
                   const float* __restrict__ bias, void* __restrict__ Dv)
{
    constexpr int NT = N / 16;                 // n-tiles
    const int lane = threadIdx.x & 31;
    const int wave = threadIdx.x >> 5;
    const int m0  = blockIdx.y * 64 + (wave >> 1) * 16;
    const int nt0 = blockIdx.x * 4 + (wave & 1) * 2;   // first of 2 n-tiles
    const int l15 = lane & 15;

    v8f c0 = {}, c1 = {};
    const _Float16* Ar = A + (size_t)(m0 + l15) * K;   // one M-row per lane
    const int ka = (lane < 16) ? 0 : 8;                // A: K pair split

#pragma unroll
    for (int kt = 0; kt < K / 32; ++kt) {
        const int k0 = kt * 32;
        AFrag af;
#pragma unroll
        for (int i = 0; i < 8; ++i) {
            // VGPR i holds K pair {0,2,4,6,16,18,20,22}[i] (+ka) per ISA layout
            const int kk = k0 + ka + ((i < 4) ? (2 * i) : (16 + 2 * (i - 4)));
            af.u[i] = *(const unsigned*)(Ar + kk);     // vectorizes to 2x b128
        }
        const _Float16* bp = Bp + ((size_t)kt * NT + nt0) * 512 + (size_t)lane * 16;
        const v16h b0 = *(const v16h*)bp;              // 32B contiguous
        const v16h b1 = *(const v16h*)(bp + 512);
        c0 = __builtin_amdgcn_wmma_f32_16x16x32_f16(false, af.v, false, b0,
                                                    (short)0, c0, false, false);
        c1 = __builtin_amdgcn_wmma_f32_16x16x32_f16(false, af.v, false, b1,
                                                    (short)0, c1, false, false);
    }

    const int n_a = nt0 * 16 + l15;
    const int n_b = n_a + 16;
    const int mb  = m0 + ((lane < 16) ? 0 : 8);
    const float bv0 = HAS_BIAS ? bias[n_a] : 0.0f;
    const float bv1 = HAS_BIAS ? bias[n_b] : 0.0f;
#pragma unroll
    for (int i = 0; i < 8; ++i) {
        float v0 = c0[i] + bv0;
        float v1 = c1[i] + bv1;
        if (RELU) { v0 = fmaxf(v0, 0.0f); v1 = fmaxf(v1, 0.0f); }
        const size_t r = (size_t)(mb + i) * N;
        if (OUTF16) {
            ((_Float16*)Dv)[r + n_a] = (_Float16)v0;
            ((_Float16*)Dv)[r + n_b] = (_Float16)v1;
        } else {
            ((float*)Dv)[r + n_a] = v0;
            ((float*)Dv)[r + n_b] = v1;
        }
    }
}

// ---------------------------------------------------------------------------
// Elementwise prep kernels
// ---------------------------------------------------------------------------
__global__ __launch_bounds__(256)
void prep_q_kernel(const float* __restrict__ lidar, const float* __restrict__ qpose,
                   _Float16* __restrict__ q16)
{
    const unsigned t = blockIdx.x * 256u + threadIdx.x;   // over MQ*256
    const int c = t & 255;
    const int r = t >> 8;             // r = b*NQ + n
    const int b = r / NQ, n = r & (NQ - 1);
    const size_t s = ((size_t)n * BS + b) * EMBED + c;
    q16[t] = (_Float16)(lidar[s] + qpose[s]);
}

__global__ __launch_bounds__(256)
void prep_feat_kernel(const float* __restrict__ feat, _Float16* __restrict__ f16)
{
    const unsigned t = blockIdx.x * 256u + threadIdx.x;   // over MV*256
    const int c = t & 255;
    const int r = t >> 8;             // r = b*NV + n
    const int b = r / NV, n = r % NV;
    const size_t s = ((size_t)n * BS + b) * EMBED + c;
    f16[t] = (_Float16)feat[s];
}

// Repack fp32 weight (K x N, row-major) into f16 fragment-packed layout (see
// GEMM comment). One output element per thread; weights are L2-trivial.
__global__ __launch_bounds__(256)
void pack_b_kernel(const float* __restrict__ W, _Float16* __restrict__ Bp, int N)
{
    const unsigned t = blockIdx.x * 256u + threadIdx.x;   // over K*N
    const int j    = t & 15;
    const int lane = (t >> 4) & 31;
    const unsigned rest = t >> 9;
    const int NT = N >> 4;
    const int nt = rest % NT;
    const int kt = rest / NT;
    const int n = nt * 16 + (lane & 15);
    const int k = kt * 32 + ((lane < 16) ? 0 : 8) + ((j < 8) ? j : (j + 8));
    Bp[t] = (_Float16)W[(size_t)k * N + n];
}

// ---------------------------------------------------------------------------
// Multi-scale deformable sampling + softmax + head accumulate.
// One wave per (b, q); lane = head-dim channel (HD == warp width == 32).
// Each corner gather reads a lane-contiguous 128B run of v[b, row, h*32+lane].
// ---------------------------------------------------------------------------
__global__ __launch_bounds__(256)
void ms_deform_sample(const float* __restrict__ vraw, const float* __restrict__ off,
                      const float* __restrict__ awr,  const float* __restrict__ ref,
                      _Float16* __restrict__ out16)
{
    const int lane = threadIdx.x & 31;
    const int wid  = (blockIdx.x << 3) + (threadIdx.x >> 5);  // b*NQ + q
    const int b = wid / NQ;
    const int q = wid & (NQ - 1);

    constexpr int Wl[4]  = {128, 64, 32, 16};
    constexpr int Hl[4]  = {128, 64, 32, 16};
    constexpr int LSI[4] = {0, 16384, 20480, 21504};

    const float* rp = ref + (size_t)(b * NQ + q) * LEVELS * 2;
    const float* op = off + (size_t)(b * NQ + q) * 256;
    const float* ap = awr + (size_t)(b * NQ + q) * 128;
    __builtin_prefetch(op, 0, 0);          // global_prefetch_b8: pull the whole
    __builtin_prefetch(ap, 0, 0);          // per-query offset/logit rows early

    float rx[4], ry[4];
#pragma unroll
    for (int l = 0; l < 4; ++l) { rx[l] = rp[2 * l]; ry[l] = rp[2 * l + 1]; }

    const size_t vb = (size_t)b * NV * EMBED + lane;
    _Float16* outp = out16 + (size_t)(b * NQ + q) * 256 + lane;

    for (int h = 0; h < HEADS; ++h) {
        // softmax over LEVELS*POINTS = 16 logits (redundant per lane, L2-hot)
        float lg[16]; float mx = -1e30f;
#pragma unroll
        for (int j = 0; j < 16; ++j) { lg[j] = ap[h * 16 + j]; mx = fmaxf(mx, lg[j]); }
        float s = 0.f;
#pragma unroll
        for (int j = 0; j < 16; ++j) { lg[j] = __expf(lg[j] - mx); s += lg[j]; }
        const float inv = 1.f / s;

        float acc = 0.f;
#pragma unroll
        for (int l = 0; l < 4; ++l) {
            const int   W  = Wl[l],        H  = Hl[l];
            const float Wf = (float)W,     Hf = (float)H;
            const size_t vlb = vb + (size_t)LSI[l] * EMBED + h * HD;
            for (int p = 0; p < 4; ++p) {
                const int   oi = ((h * 4 + l) * 4 + p) * 2;
                const float lx = rx[l] + op[oi]     / Wf;
                const float ly = ry[l] + op[oi + 1] / Hf;
                const float xx = lx * Wf - 0.5f;
                const float yy = ly * Hf - 0.5f;
                const float x0f = floorf(xx), y0f = floorf(yy);
                const float fx = xx - x0f,    fy = yy - y0f;
                const int   x0 = (int)x0f,    y0 = (int)y0f;
                const float wq = lg[l * 4 + p] * inv;

                auto corner = [&](int xi, int yi, float w) {
                    const bool valid = (xi >= 0) && (xi < W) && (yi >= 0) && (yi < H);
                    const int xc = xi < 0 ? 0 : (xi > W - 1 ? W - 1 : xi);
                    const int yc = yi < 0 ? 0 : (yi > H - 1 ? H - 1 : yi);
                    const float g = vraw[vlb + (size_t)(yc * W + xc) * EMBED];
                    acc += valid ? (wq * w * g) : 0.f;
                };
                corner(x0,     y0,     (1.f - fx) * (1.f - fy));
                corner(x0 + 1, y0,     fx * (1.f - fy));
                corner(x0,     y0 + 1, (1.f - fx) * fy);
                corner(x0 + 1, y0 + 1, fx * fy);
            }
        }
        outp[h * HD] = (_Float16)acc;
    }
}

// ---------------------------------------------------------------------------
// LayerNorm kernels: one wave per row of 256 channels (8 vals/lane + shfl tree)
// ---------------------------------------------------------------------------
__global__ __launch_bounds__(256)
void ln1_kernel(const float* __restrict__ y, const float* __restrict__ lidar,
                const float* __restrict__ g, const float* __restrict__ bb,
                float* __restrict__ xf, _Float16* __restrict__ x16)
{
    const int lane = threadIdx.x & 31;
    const int row  = (blockIdx.x << 3) + (threadIdx.x >> 5);  // b*NQ + n
    const int b = row / NQ, n = row & (NQ - 1);
    float v[8]; float s1 = 0.f;
#pragma unroll
    for (int i = 0; i < 8; ++i) {
        const int c = i * 32 + lane;
        v[i] = y[(size_t)row * 256 + c] + lidar[((size_t)n * BS + b) * 256 + c];
        s1 += v[i];
    }
    for (int o = 16; o > 0; o >>= 1) s1 += __shfl_xor(s1, o, 32);
    const float mu = s1 * (1.f / 256.f);
    float s2 = 0.f;
#pragma unroll
    for (int i = 0; i < 8; ++i) { const float d = v[i] - mu; s2 += d * d; }
    for (int o = 16; o > 0; o >>= 1) s2 += __shfl_xor(s2, o, 32);
    const float rstd = rsqrtf(s2 * (1.f / 256.f) + 1e-5f);
#pragma unroll
    for (int i = 0; i < 8; ++i) {
        const int c = i * 32 + lane;
        const float hv = (v[i] - mu) * rstd * g[c] + bb[c];
        xf [(size_t)row * 256 + c] = hv;
        x16[(size_t)row * 256 + c] = (_Float16)hv;
    }
}

__global__ __launch_bounds__(256)
void ln2_kernel(const float* __restrict__ z, const float* __restrict__ xf,
                const float* __restrict__ g, const float* __restrict__ bb,
                float* __restrict__ outp)
{
    const int lane = threadIdx.x & 31;
    const int row  = (blockIdx.x << 3) + (threadIdx.x >> 5);  // b*NQ + n
    const int b = row / NQ, n = row & (NQ - 1);
    float v[8]; float s1 = 0.f;
#pragma unroll
    for (int i = 0; i < 8; ++i) {
        const int c = i * 32 + lane;
        v[i] = z[(size_t)row * 256 + c] + xf[(size_t)row * 256 + c];
        s1 += v[i];
    }
    for (int o = 16; o > 0; o >>= 1) s1 += __shfl_xor(s1, o, 32);
    const float mu = s1 * (1.f / 256.f);
    float s2 = 0.f;
#pragma unroll
    for (int i = 0; i < 8; ++i) { const float d = v[i] - mu; s2 += d * d; }
    for (int o = 16; o > 0; o >>= 1) s2 += __shfl_xor(s2, o, 32);
    const float rstd = rsqrtf(s2 * (1.f / 256.f) + 1e-5f);
#pragma unroll
    for (int i = 0; i < 8; ++i) {
        const int c = i * 32 + lane;
        outp[((size_t)n * BS + b) * 256 + c] = (v[i] - mu) * rstd * g[c] + bb[c];
    }
}

// ---------------------------------------------------------------------------
// Host-side launch (graph-capture safe: only kernel launches on `stream`)
// ---------------------------------------------------------------------------
extern "C" void kernel_launch(void* const* d_in, const int* in_sizes, int n_in,
                              void* d_out, int out_size, void* d_ws, size_t ws_size,
                              hipStream_t stream)
{
    (void)in_sizes; (void)n_in; (void)out_size; (void)ws_size;

    const float* lidar  = (const float*)d_in[0];
    const float* feat   = (const float*)d_in[1];
    const float* ref    = (const float*)d_in[2];
    const float* qpose  = (const float*)d_in[5];
    const float* W_off  = (const float*)d_in[6];
    const float* b_off  = (const float*)d_in[7];
    const float* W_attn = (const float*)d_in[8];
    const float* b_attn = (const float*)d_in[9];
    const float* W_val  = (const float*)d_in[10];
    const float* b_val  = (const float*)d_in[11];
    const float* W_out  = (const float*)d_in[12];
    const float* b_out  = (const float*)d_in[13];
    const float* ln1g   = (const float*)d_in[14];
    const float* ln1b   = (const float*)d_in[15];
    const float* ffn1   = (const float*)d_in[16];
    const float* ffn2   = (const float*)d_in[17];
    const float* ln2g   = (const float*)d_in[18];
    const float* ln2b   = (const float*)d_in[19];

    // Workspace layout with phase-based aliasing (all offsets 256B aligned).
    constexpr size_t OFF_Q16    = 0;                         // 16.8 MB (later: y f32)
    constexpr size_t OFF_FEAT16 = 16777216;                  // 22.3 MB (part of y region)
    constexpr size_t OFF_V      = 39059456;                  // 44.6 MB (later: xf f32)
    constexpr size_t OFF_OFF    = 83623936;                  // 33.5 MB (later: z f32)
    constexpr size_t OFF_AWR    = 117178368;                 // 16.8 MB (later: x16 f16)
    constexpr size_t OFF_OUT16  = 133955584;                 // 16.8 MB
    constexpr size_t OFF_T16    = 150732800;                 // 16.8 MB
    constexpr size_t OFF_W16    = 167510016;                 // 0.7 MB packed-f16 weights

    char* ws = (char*)d_ws;
    _Float16* q16    = (_Float16*)(ws + OFF_Q16);
    _Float16* feat16 = (_Float16*)(ws + OFF_FEAT16);
    float*    vraw   = (float*)   (ws + OFF_V);
    float*    offb   = (float*)   (ws + OFF_OFF);
    float*    awr    = (float*)   (ws + OFF_AWR);
    _Float16* out16  = (_Float16*)(ws + OFF_OUT16);
    _Float16* t16    = (_Float16*)(ws + OFF_T16);
    _Float16* wv16   = (_Float16*)(ws + OFF_W16);
    _Float16* woff16 = (_Float16*)(ws + OFF_W16 + 131072);
    _Float16* wattn16= (_Float16*)(ws + OFF_W16 + 262144);
    _Float16* wout16 = (_Float16*)(ws + OFF_W16 + 327680);
    _Float16* f116   = (_Float16*)(ws + OFF_W16 + 458752);
    _Float16* f216   = (_Float16*)(ws + OFF_W16 + 589824);
    // aliased (later phases)
    float*    y      = (float*)   (ws + OFF_Q16);   // over q16+feat16 (dead)
    float*    xf     = (float*)   (ws + OFF_V);     // over vraw (dead)
    _Float16* x16    = (_Float16*)(ws + OFF_AWR);   // over awr (dead)
    float*    z      = (float*)   (ws + OFF_OFF);   // over offb (dead)

    // --- phase 1: transpose + f16 conversion + B fragment packing ----------
    prep_q_kernel   <<<MQ, 256, 0, stream>>>(lidar, qpose, q16);
    prep_feat_kernel<<<MV, 256, 0, stream>>>(feat, feat16);
    pack_b_kernel<<<65536 / 256, 256, 0, stream>>>(W_val,  wv16,    256);
    pack_b_kernel<<<65536 / 256, 256, 0, stream>>>(W_off,  woff16,  256);
    pack_b_kernel<<<32768 / 256, 256, 0, stream>>>(W_attn, wattn16, 128);
    pack_b_kernel<<<65536 / 256, 256, 0, stream>>>(W_out,  wout16,  256);
    pack_b_kernel<<<65536 / 256, 256, 0, stream>>>(ffn1,   f116,    256);
    pack_b_kernel<<<65536 / 256, 256, 0, stream>>>(ffn2,   f216,    256);

    // --- phase 2: projection GEMMs (WMMA f16 -> f32 acc) -------------------
    gemm_f16_wmma<1,0,0,256,256><<<dim3(4, MV/64), 256, 0, stream>>>(feat16, wv16,   b_val,  vraw);
    gemm_f16_wmma<1,0,0,256,256><<<dim3(4, MQ/64), 256, 0, stream>>>(q16,    woff16, b_off,  offb);
    gemm_f16_wmma<1,0,0,128,256><<<dim3(2, MQ/64), 256, 0, stream>>>(q16,    wattn16,b_attn, awr);

    // --- phase 3: deformable sampling + softmax ----------------------------
    ms_deform_sample<<<MQ / 8, 256, 0, stream>>>(vraw, offb, awr, ref, out16);

    // --- phase 4: output proj + residual/LN + FFN + LN ---------------------
    gemm_f16_wmma<1,0,0,256,256><<<dim3(4, MQ/64), 256, 0, stream>>>(out16, wout16, b_out, y);
    ln1_kernel<<<MQ / 8, 256, 0, stream>>>(y, lidar, ln1g, ln1b, xf, x16);
    gemm_f16_wmma<0,1,1,256,256><<<dim3(4, MQ/64), 256, 0, stream>>>(x16, f116, nullptr, t16);
    gemm_f16_wmma<0,0,0,256,256><<<dim3(4, MQ/64), 256, 0, stream>>>(t16, f216, nullptr, z);
    ln2_kernel<<<MQ / 8, 256, 0, stream>>>(z, xf, ln2g, ln2b, (float*)d_out);
}
